// SelfAttention_26920855011406
// MI455X (gfx1250) — matrix operand order repaired and checked by reading, not compiled, not run on hardware
//
#include <hip/hip_runtime.h>
#include <stdint.h>

typedef __attribute__((ext_vector_type(16))) __bf16        v16bf;
typedef __attribute__((ext_vector_type(8)))  float         v8f;
typedef __attribute__((ext_vector_type(4)))  unsigned int  u32x4;

#define N_PIX 4096
#define C_IN  256
#define C_OUT 32
#define NCH   (N_PIX / 32)      // 128 key chunks of 32
#define KROW  40                // padded LDS row stride in ushorts (80 B)

// f32 -> bf16 (RNE) through the hardware convert path: fptrunc float->bfloat
// lowers to native cvt on gfx1250 (and is generically legalized otherwise).
static __device__ __forceinline__ unsigned short f2bf(float f) {
    union { __bf16 h; unsigned short u; } cv;
    cv.h = (__bf16)f;
    return cv.u;
}

union AFrag {
    v16bf v;
    u32x4 q[2];
    unsigned short u[16];
};

// one 16-byte async DMA: global -> LDS, tracked by ASYNCcnt
static __device__ __forceinline__ void async_cp_b128(unsigned lds_off, const void* gptr) {
    asm volatile("global_load_async_to_lds_b128 %0, %1, off"
                 :: "v"(lds_off), "v"((unsigned long long)(uintptr_t)gptr)
                 : "memory");
}
#define ASYNC_WAIT(n) asm volatile("s_wait_asynccnt " #n ::: "memory")

// ---------------------------------------------------------------------------
// Projection: y = W @ x + b per batch.  A = W tile (16o x 32c), B = x tile
// (32c x 16n), 8 K-steps over C_IN=256.  f/g stored transposed [N,32] bf16,
// h stored [256,N] bf16.
// ---------------------------------------------------------------------------
__global__ __launch_bounds__(256) void proj_kernel(
        const float* __restrict__ x, const float* __restrict__ W,
        const float* __restrict__ bias, unsigned short* __restrict__ out,
        int O, int transposed)
{
    const int b    = blockIdx.y;
    const int wave = threadIdx.x >> 5;
    const int lane = threadIdx.x & 31;
    const int hl   = lane >> 4;
    const int l16  = lane & 15;

    const int oTiles = O >> 4;
    const int t      = blockIdx.x * 8 + wave;
    const int obase  = (t % oTiles) * 16;
    const int nbase  = (t / oTiles) * 16;

    const float* xb = x + (size_t)b * C_IN * N_PIX;

    v8f acc = {};
    const int orow = obase + l16;
    const int ncol = nbase + l16;

    for (int c0 = 0; c0 < C_IN; c0 += 32) {
        AFrag a;
        #pragma unroll
        for (int e = 0; e < 8; ++e) {
            a.u[e]     = f2bf(W[(size_t)orow * C_IN + c0 + 8 * hl + e]);
            a.u[e + 8] = f2bf(W[(size_t)orow * C_IN + c0 + 16 + 8 * hl + e]);
        }
        AFrag bf_;
        #pragma unroll
        for (int e = 0; e < 16; ++e)
            bf_.u[e] = f2bf(xb[(size_t)(c0 + (e & 7) + 8 * hl + 16 * (e >> 3)) * N_PIX + ncol]);

        acc = __builtin_amdgcn_wmma_f32_16x16x32_bf16(false, a.v, false, bf_.v,
                                                      (short)0, acc, false, false);
    }
    #pragma unroll
    for (int e = 0; e < 8; ++e)
        acc[e] += bias[obase + 8 * hl + e];

    if (transposed) {
        union { u32x4 q; unsigned short s[8]; } pk;
        #pragma unroll
        for (int e = 0; e < 8; ++e) pk.s[e] = f2bf(acc[e]);
        size_t idx = (size_t)b * N_PIX * O + (size_t)ncol * O + obase + 8 * hl;
        *(u32x4*)(out + idx) = pk.q;
    } else {
        unsigned short* ob = out + (size_t)b * O * N_PIX;
        #pragma unroll
        for (int e = 0; e < 8; ++e)
            ob[(size_t)(obase + 8 * hl + e) * N_PIX + ncol] = f2bf(acc[e]);
    }
}

// ---------------------------------------------------------------------------
// Fused flash attention with double-buffered async global->LDS staging.
// 4 waves / block, each wave owns 16 queries; all waves share the staged
// K-chunk (32x32 bf16) and V-chunk (256x32 bf16) per 32-key iteration.
// ---------------------------------------------------------------------------
__global__ __launch_bounds__(128) void attn_kernel(
        const unsigned short* __restrict__ fT,   // [b][N][32]  keys
        const unsigned short* __restrict__ gT,   // [b][N][32]  queries
        const unsigned short* __restrict__ hN,   // [b][256][N] values
        float* __restrict__ out)                 // [b][256][N]
{
    __shared__ unsigned short ldsK[2][32 * KROW];    // 2 x 2.5 KB
    __shared__ unsigned short ldsV[2][256 * KROW];   // 2 x 20 KB

    const int b    = blockIdx.y;
    const int t    = threadIdx.x;
    const int wave = t >> 5;
    const int lane = t & 31;
    const int hl   = lane >> 4;
    const int l16  = lane & 15;

    const int qm = blockIdx.x * 64 + wave * 16 + l16;

    const unsigned short* fb = fT + (size_t)b * N_PIX * C_OUT;
    const unsigned short* gb = gT + (size_t)b * N_PIX * C_OUT;
    const unsigned short* hb = hN + (size_t)b * C_IN * N_PIX;

    // Q as B-type fragment: elem e -> channel c = (e&7) + 8*hl + 16*(e>>3), col = qm
    AFrag qf;
    {
        const unsigned short* p = gb + (size_t)qm * C_OUT;
        qf.q[0] = *(const u32x4*)(p + 8 * hl);
        qf.q[1] = *(const u32x4*)(p + 16 + 8 * hl);
    }

    // issue one chunk's DMA: 1 K-transfer + 8 V-transfers per lane = 9/wave
    auto issue = [&](int nb, int bufn) {
        {
            const int row = t >> 2, seg = t & 3;             // 128 x 16B = 2 KB
            unsigned loff = (unsigned)(size_t)&ldsK[bufn][row * KROW + seg * 8];
            async_cp_b128(loff, fb + (size_t)(nb + row) * C_OUT + seg * 8);
        }
        #pragma unroll
        for (int i = 0; i < 8; ++i) {                        // 1024 x 16B = 16 KB
            const int j = i * 128 + t;
            const int c = j >> 2, seg = j & 3;
            unsigned loff = (unsigned)(size_t)&ldsV[bufn][c * KROW + seg * 8];
            async_cp_b128(loff, hb + (size_t)c * N_PIX + nb + seg * 8);
        }
    };

    v8f o_acc[16] = {};
    float run_max = -1e30f;
    float run_sum = 0.0f;

    issue(0, 0);
    issue(32, 1);

    for (int i = 0; i < NCH; ++i) {
        if (i + 1 < NCH) { ASYNC_WAIT(9); } else { ASYNC_WAIT(0); }
        __syncthreads();                       // chunk i resident for all waves
        const int buf = i & 1;

        // K A-fragments for keys [0,16) and [16,32) of the chunk
        AFrag k0, k1;
        {
            const unsigned short* p0 = &ldsK[buf][l16 * KROW];
            k0.q[0] = *(const u32x4*)(p0 + 8 * hl);
            k0.q[1] = *(const u32x4*)(p0 + 16 + 8 * hl);
            const unsigned short* p1 = &ldsK[buf][(16 + l16) * KROW];
            k1.q[0] = *(const u32x4*)(p1 + 8 * hl);
            k1.q[1] = *(const u32x4*)(p1 + 16 + 8 * hl);
        }
        v8f z = {};
        v8f s0 = __builtin_amdgcn_wmma_f32_16x16x32_bf16(false, k0.v, false, qf.v,
                                                         (short)0, z, false, false);
        v8f s1 = __builtin_amdgcn_wmma_f32_16x16x32_bf16(false, k1.v, false, qf.v,
                                                         (short)0, z, false, false);
        // S^T layout: row = key (e + 8*hl per tile), col = query (lane%16)
        float cm = -1e30f;
        #pragma unroll
        for (int e = 0; e < 8; ++e) cm = fmaxf(cm, fmaxf(s0[e], s1[e]));
        cm = fmaxf(cm, __shfl_xor(cm, 16, 32));
        const float nm   = fmaxf(run_max, cm);
        const float corr = __expf(run_max - nm);

        float p0v[8], p1v[8];
        float csum = 0.0f;
        #pragma unroll
        for (int e = 0; e < 8; ++e) {
            p0v[e] = __expf(s0[e] - nm);
            p1v[e] = __expf(s1[e] - nm);
            csum += p0v[e] + p1v[e];
        }
        csum += __shfl_xor(csum, 16, 32);
        run_sum = run_sum * corr + csum;
        run_max = nm;

        #pragma unroll
        for (int f = 0; f < 16; ++f)
            #pragma unroll
            for (int e = 0; e < 8; ++e) o_acc[f][e] *= corr;

        // P as B-type fragment (32 keys x 16 queries): elem e -> key = e + 16*hl
        AFrag pb;
        #pragma unroll
        for (int e = 0; e < 8; ++e) {
            float oth0 = __shfl_xor(p0v[e], 16, 32);
            float oth1 = __shfl_xor(p1v[e], 16, 32);
            float lo = hl ? oth1 : p0v[e];
            float hi = hl ? p1v[e] : oth0;
            pb.u[e]     = f2bf(lo);
            pb.u[e + 8] = f2bf(hi);
        }

        // O^T[c][m] += V[c][keys] * P[keys][m], 16 channel tiles from LDS
        #pragma unroll
        for (int ct = 0; ct < 16; ++ct) {
            AFrag va;
            const unsigned short* p = &ldsV[buf][(ct * 16 + l16) * KROW];
            va.q[0] = *(const u32x4*)(p + 16 * hl);
            va.q[1] = *(const u32x4*)(p + 16 * hl + 8);
            o_acc[ct] = __builtin_amdgcn_wmma_f32_16x16x32_bf16(false, va.v, false, pb.v,
                                                                (short)0, o_acc[ct],
                                                                false, false);
        }

        __syncthreads();                       // all waves done reading buf
        if (i + 2 < NCH) issue((i + 2) * 32, buf);
    }

    const float inv = 1.0f / run_sum;
    float* ob = out + (size_t)b * C_IN * N_PIX;
    #pragma unroll
    for (int ct = 0; ct < 16; ++ct)
        #pragma unroll
        for (int e = 0; e < 8; ++e)
            ob[(size_t)(ct * 16 + 8 * hl + e) * N_PIX + qm] = o_acc[ct][e] * inv;
}

// ---------------------------------------------------------------------------
extern "C" void kernel_launch(void* const* d_in, const int* in_sizes, int n_in,
                              void* d_out, int out_size, void* d_ws, size_t ws_size,
                              hipStream_t stream)
{
    (void)in_sizes; (void)n_in; (void)out_size; (void)ws_size;
    const float* x   = (const float*)d_in[0];
    const float* Wf  = (const float*)d_in[1];
    const float* bfv = (const float*)d_in[2];
    const float* Wg  = (const float*)d_in[3];
    const float* bgv = (const float*)d_in[4];
    const float* Wh  = (const float*)d_in[5];
    const float* bhv = (const float*)d_in[6];
    float* out = (float*)d_out;

    unsigned short* fT = (unsigned short*)d_ws;                 // 4*4096*32 bf16
    unsigned short* gT = fT + (size_t)4 * N_PIX * C_OUT;        // 4*4096*32 bf16
    unsigned short* hB = gT + (size_t)4 * N_PIX * C_OUT;        // 4*256*4096 bf16

    proj_kernel<<<dim3(64, 4),  dim3(256), 0, stream>>>(x, Wf, bfv, fT, C_OUT, 1);
    proj_kernel<<<dim3(64, 4),  dim3(256), 0, stream>>>(x, Wg, bgv, gT, C_OUT, 1);
    proj_kernel<<<dim3(512, 4), dim3(256), 0, stream>>>(x, Wh, bhv, hB, C_IN, 0);
    attn_kernel<<<dim3(N_PIX / 64, 4), dim3(128), 0, stream>>>(fT, gT, hB, out);
}
